// EGConvEdgeClassifier_86938728005820
// MI455X (gfx1250) — matrix operand order repaired and checked by reading, not compiled, not run on hardware
//
#include <hip/hip_runtime.h>

typedef _Float16 f16;
typedef __attribute__((ext_vector_type(16))) _Float16 v16h;
typedef __attribute__((ext_vector_type(8)))  _Float16 v8h;
typedef __attribute__((ext_vector_type(8)))  float    v8f;

#define HID 128   // hidden channels (= IN)
#define BFH 64    // BASES * FH
#define HB  32    // HEADS * BASES
#define WCOLS 96  // BFH + HB, columns of fused weight matrix
#define WLD 136   // padded LDS row pitch (halves): 16B aligned, bank-conflict friendly

// ---------------------------------------------------------------- utilities
__global__ void zero_f32(float* __restrict__ p, int n) {
  int i = blockIdx.x * blockDim.x + threadIdx.x;
  int stride = gridDim.x * blockDim.x;
  for (; i < n; i += stride) p[i] = 0.0f;
}

// deg[c] += 1 for every edge (incl. self loops)
__global__ void degree_kernel(const int* __restrict__ ei, int E, int N,
                              float* __restrict__ deg) {
  int i = blockIdx.x * blockDim.x + threadIdx.x;
  int total = E + N;
  if (i >= total) return;
  int c = (i < E) ? ei[E + i] : (i - E);
  atomicAdd(deg + c, 1.0f);
}

// dis[i] = deg>0 ? deg^-0.5 : 0   (in place)
__global__ void finalize_dis(float* __restrict__ dis, int N) {
  int i = blockIdx.x * blockDim.x + threadIdx.x;
  if (i >= N) return;
  float d = dis[i];
  dis[i] = (d > 0.0f) ? rsqrtf(d) : 0.0f;
}

// ------------------------------------------------- fused WMMA GEMM (per layer)
// Block = 128 threads = 4 wave32; each wave owns one 16-row tile and computes
// all 6 column tiles (4 -> bases[N,64], 2 -> wout[N,32]+bc).
// Weights [Wb|Wc] (128x96 f32) are staged once per block into LDS as f16,
// TRANSPOSED (Wt[n][k], pitch WLD) so each B fragment is 2 contiguous
// ds_load_b128 instead of 16 strided global loads.
__global__ void __launch_bounds__(128)
gemm_wmma(const float* __restrict__ A, const float* __restrict__ Wb,
          const float* __restrict__ Wc, const float* __restrict__ bc,
          float* __restrict__ bases, float* __restrict__ wout, int rowTiles) {
  __shared__ f16 Wt[WCOLS * WLD];

  // ---- cooperative load: global f32 (coalesced along n) -> LDS f16 transposed
  for (int i = threadIdx.x; i < HID * WCOLS; i += blockDim.x) {
    int k = i / WCOLS;          // 0..127
    int n = i % WCOLS;          // 0..95
    float v = (n < BFH) ? Wb[k * BFH + n] : Wc[k * HB + (n - BFH)];
    Wt[n * WLD + k] = (f16)v;
  }
  __syncthreads();

  const int wave = threadIdx.x >> 5;
  const int rt   = blockIdx.x * 4 + wave;
  if (rt >= rowTiles) return;                 // wave-uniform: EXEC stays all-1s

  const int  lane   = threadIdx.x & 31;
  const int  lo     = lane & 15;
  const bool hiHalf = lane >= 16;
  const int  m      = rt * 16 + lo;           // my A row

  v8f acc[6] = {};
  #pragma unroll
  for (int kc = 0; kc < 4; ++kc) {
    // ---- A fragment (16x32 f16): lanes<16 hold K {0..7,16..23}, lanes>=16 {8..15,24..31}
    const int ka = kc * 32 + (hiHalf ? 8 : 0);
    const float* arow = A + (long long)m * HID + ka;
    const float4 a0 = *(const float4*)(arow + 0);
    const float4 a1 = *(const float4*)(arow + 4);
    const float4 a2 = *(const float4*)(arow + 16);
    const float4 a3 = *(const float4*)(arow + 20);
    v16h a;
    a[0]=(f16)a0.x; a[1]=(f16)a0.y; a[2]=(f16)a0.z; a[3]=(f16)a0.w;
    a[4]=(f16)a1.x; a[5]=(f16)a1.y; a[6]=(f16)a1.z; a[7]=(f16)a1.w;
    a[8]=(f16)a2.x; a[9]=(f16)a2.y; a[10]=(f16)a2.z; a[11]=(f16)a2.w;
    a[12]=(f16)a3.x; a[13]=(f16)a3.y; a[14]=(f16)a3.z; a[15]=(f16)a3.w;

    // ---- B fragments from LDS: lane n<16 holds col n K {0..7,16..23}; lanes>=16 {8..15,24..31}
    const int kb = kc * 32 + (hiHalf ? 8 : 0);
    #pragma unroll
    for (int t = 0; t < 6; ++t) {
      const f16* wrow = &Wt[(t * 16 + lo) * WLD + kb];
      const v8h b0 = *(const v8h*)(wrow);       // ds_load_b128
      const v8h b1 = *(const v8h*)(wrow + 16);  // ds_load_b128
      v16h b;
      #pragma unroll
      for (int i = 0; i < 8; ++i) { b[i] = b0[i]; b[i + 8] = b1[i]; }
      acc[t] = __builtin_amdgcn_wmma_f32_16x16x32_f16(
          /*neg_a=*/false, a, /*neg_b=*/false, b,
          /*c_mod=*/(short)0, acc[t], /*reuse_a=*/false, /*reuse_b=*/false);
    }
  }

  // ---- store C/D: VGPR i = row i (lanes<16) or row 8+i (lanes>=16), col = lane&15
  const int mrow0 = rt * 16 + (hiHalf ? 8 : 0);
  #pragma unroll
  for (int t = 0; t < 4; ++t) {
    const int n = t * 16 + lo;
    #pragma unroll
    for (int i = 0; i < 8; ++i)
      bases[(long long)(mrow0 + i) * BFH + n] = acc[t][i];
  }
  #pragma unroll
  for (int t = 4; t < 6; ++t) {
    const int n = (t - 4) * 16 + lo;
    const float bias = bc[n];
    #pragma unroll
    for (int i = 0; i < 8; ++i)
      wout[(long long)(mrow0 + i) * HB + n] = acc[t][i] + bias;
  }
}

// ------------------------------------- symnorm gather-scale-scatter (atomics)
// One thread per (edge, 4-float chunk): agg[col] += bases[row] * dis[row]*dis[col]
__global__ void scatter_kernel(const float* __restrict__ bases,
                               const float* __restrict__ dis,
                               const int* __restrict__ ei, int E, int N,
                               float* __restrict__ agg) {
  long long idx = (long long)blockIdx.x * blockDim.x + threadIdx.x;
  int e = (int)(idx >> 4);
  if (e >= E + N) return;
  int f = ((int)idx & 15) * 4;
  int r, c;
  if (e < E) { r = ei[e]; c = ei[E + e]; } else { r = e - E; c = r; }
  float nw = dis[r] * dis[c];
  const float4 v = *(const float4*)(bases + (long long)r * BFH + f);
  float* dst = agg + (long long)c * BFH + f;
  atomicAdd(dst + 0, v.x * nw);
  atomicAdd(dst + 1, v.y * nw);
  atomicAdd(dst + 2, v.z * nw);
  atomicAdd(dst + 3, v.w * nw);
}

// ----------------------------- per-node head/basis combine: out = w x agg + b
__global__ void combine_kernel(const float* __restrict__ w,
                               const float* __restrict__ agg,
                               const float* __restrict__ bias,
                               float* __restrict__ out, int N, int relu) {
  int idx = blockIdx.x * blockDim.x + threadIdx.x;
  if (idx >= N * HID) return;
  int n = idx >> 7, o = idx & 127;
  int hh = o >> 4, f = o & 15;
  const float* wn = w   + (long long)n * HB  + hh * 4;  // [h][b]
  const float* an = agg + (long long)n * BFH + f;       // [b][f]
  float s = bias[o];
  #pragma unroll
  for (int b = 0; b < 4; ++b) s = fmaf(wn[b], an[b * 16], s);
  if (relu) s = fmaxf(s, 0.0f);
  out[idx] = s;
}

// ------------------------------------------ edge classifier on ORIGINAL edges
__global__ void classifier_kernel(const float* __restrict__ h,
                                  const int* __restrict__ ei,
                                  const float* __restrict__ Wcls,
                                  const float* __restrict__ bcls,
                                  float* __restrict__ out, int E) {
  int e = blockIdx.x * blockDim.x + threadIdx.x;
  if (e >= E) return;
  int r = ei[e], c = ei[E + e];
  const float* hr = h + (long long)r * HID;
  const float* hc = h + (long long)c * HID;
  float s0 = bcls[0], s1 = bcls[1];
  #pragma unroll 4
  for (int k4 = 0; k4 < HID; k4 += 4) {
    const float4 vr = *(const float4*)(hr + k4);
    const float4 vc = *(const float4*)(hc + k4);
    s0 = fmaf(vr.x, Wcls[(k4+0)*2+0], s0); s1 = fmaf(vr.x, Wcls[(k4+0)*2+1], s1);
    s0 = fmaf(vr.y, Wcls[(k4+1)*2+0], s0); s1 = fmaf(vr.y, Wcls[(k4+1)*2+1], s1);
    s0 = fmaf(vr.z, Wcls[(k4+2)*2+0], s0); s1 = fmaf(vr.z, Wcls[(k4+2)*2+1], s1);
    s0 = fmaf(vr.w, Wcls[(k4+3)*2+0], s0); s1 = fmaf(vr.w, Wcls[(k4+3)*2+1], s1);
    s0 = fmaf(vc.x, Wcls[(HID+k4+0)*2+0], s0); s1 = fmaf(vc.x, Wcls[(HID+k4+0)*2+1], s1);
    s0 = fmaf(vc.y, Wcls[(HID+k4+1)*2+0], s0); s1 = fmaf(vc.y, Wcls[(HID+k4+1)*2+1], s1);
    s0 = fmaf(vc.z, Wcls[(HID+k4+2)*2+0], s0); s1 = fmaf(vc.z, Wcls[(HID+k4+2)*2+1], s1);
    s0 = fmaf(vc.w, Wcls[(HID+k4+3)*2+0], s0); s1 = fmaf(vc.w, Wcls[(HID+k4+3)*2+1], s1);
  }
  out[e * 2 + 0] = s0;
  out[e * 2 + 1] = s1;
}

// ---------------------------------------------------------------- launcher
extern "C" void kernel_launch(void* const* d_in, const int* in_sizes, int n_in,
                              void* d_out, int out_size, void* d_ws, size_t ws_size,
                              hipStream_t stream) {
  const float* x    = (const float*)d_in[0];
  const int*   ei   = (const int*)  d_in[1];
  const float* Wb1  = (const float*)d_in[2];
  const float* Wc1  = (const float*)d_in[3];
  const float* bc1  = (const float*)d_in[4];
  const float* b1   = (const float*)d_in[5];
  const float* Wb2  = (const float*)d_in[6];
  const float* Wc2  = (const float*)d_in[7];
  const float* bc2  = (const float*)d_in[8];
  const float* b2   = (const float*)d_in[9];
  const float* Wcls = (const float*)d_in[10];
  const float* bcls = (const float*)d_in[11];
  float* out = (float*)d_out;

  const int N = in_sizes[0] / HID;
  const int E = in_sizes[1] / 2;

  // workspace carve-up (floats)
  float* ws    = (float*)d_ws;
  float* dis   = ws;  ws += N;
  float* bases = ws;  ws += (size_t)N * BFH;
  float* wcmb  = ws;  ws += (size_t)N * HB;
  float* agg   = ws;  ws += (size_t)N * BFH;
  float* hbuf  = ws;  ws += (size_t)N * HID;

  const int rowTiles   = (N + 15) / 16;
  const int gemmBlocks = (rowTiles + 3) / 4;
  const long long sthreads = (long long)(E + N) * 16;

  // symnorm degrees
  zero_f32<<<256, 256, 0, stream>>>(dis, N);
  degree_kernel<<<(E + N + 255) / 256, 256, 0, stream>>>(ei, E, N, dis);
  finalize_dis<<<(N + 255) / 256, 256, 0, stream>>>(dis, N);

  // ---- layer 1: x -> hbuf (relu)
  gemm_wmma<<<gemmBlocks, 128, 0, stream>>>(x, Wb1, Wc1, bc1, bases, wcmb, rowTiles);
  zero_f32<<<2048, 256, 0, stream>>>(agg, N * BFH);
  scatter_kernel<<<(int)((sthreads + 255) / 256), 256, 0, stream>>>(bases, dis, ei, E, N, agg);
  combine_kernel<<<(N * HID + 255) / 256, 256, 0, stream>>>(wcmb, agg, b1, hbuf, N, 1);

  // ---- layer 2: hbuf -> hbuf (no relu)
  gemm_wmma<<<gemmBlocks, 128, 0, stream>>>(hbuf, Wb2, Wc2, bc2, bases, wcmb, rowTiles);
  zero_f32<<<2048, 256, 0, stream>>>(agg, N * BFH);
  scatter_kernel<<<(int)((sthreads + 255) / 256), 256, 0, stream>>>(bases, dis, ei, E, N, agg);
  combine_kernel<<<(N * HID + 255) / 256, 256, 0, stream>>>(wcmb, agg, b2, hbuf, N, 0);

  // ---- edge classifier
  classifier_kernel<<<(E + 255) / 256, 256, 0, stream>>>(hbuf, ei, Wcls, bcls, out, E);
}
